// TransformerUnityLayer_16690242912691
// MI455X (gfx1250) — compile-verified, matrix operand order checked by reading, not run on hardware
//
#include <hip/hip_runtime.h>
#include <math.h>

// ---------------------------------------------------------------------------
// MI455X (gfx1250) implementation.
// All large GEMMs use v_wmma_f32_16x16x32_bf16 (wave32 WMMA), fp32 accum.
// ---------------------------------------------------------------------------

typedef __bf16 bf16_t;
typedef __attribute__((ext_vector_type(16))) __bf16 v16bf;
typedef __attribute__((ext_vector_type(8)))  __bf16 v8bf;
typedef __attribute__((ext_vector_type(8)))  float  v8f;

#define PHI_F 1.618033988749895f

// ---------------- helpers ----------------

__device__ __forceinline__ float block_reduce_sum256(float v, float* sh) {
    int tid = threadIdx.x;
    sh[tid] = v; __syncthreads();
    for (int s = 128; s > 0; s >>= 1) {
        if (tid < s) sh[tid] += sh[tid + s];
        __syncthreads();
    }
    float r = sh[0]; __syncthreads();
    return r;
}

__device__ __forceinline__ float block_reduce_max256(float v, float* sh) {
    int tid = threadIdx.x;
    sh[tid] = v; __syncthreads();
    for (int s = 128; s > 0; s >>= 1) {
        if (tid < s) sh[tid] = fmaxf(sh[tid], sh[tid + s]);
        __syncthreads();
    }
    float r = sh[0]; __syncthreads();
    return r;
}

// Load one 16-element bf16 WMMA fragment for this lane.
// Per-lane K mapping (16-bit A/B operand, 16x16x32):
//   elems 0..7  -> K = k0 + 8*lanehi + (0..7)
//   elems 8..15 -> K = k0 + 16 + 8*lanehi + (0..7)
__device__ __forceinline__ v16bf load_frag(const bf16_t* row_k0, int khi) {
    v8bf lo = *(const v8bf*)(row_k0 + khi);
    v8bf hi = *(const v8bf*)(row_k0 + khi + 16);
    v16bf f;
#pragma unroll
    for (int i = 0; i < 8; ++i) { f[i] = lo[i]; f[i + 8] = hi[i]; }
    return f;
}

// ---------------- generic bf16 WMMA GEMM ----------------
// C[m,n] = act( alpha * sum_k A[m,k]*W[n,k] + bias[n] )
// A: M x K (row-major, lda), W: N x K (row-major, ldw).
// Each wave computes a 32x32 tile (4x v_wmma_f32_16x16x32_bf16 per K step).
// store_bf16: 1 -> bf16 store, 0 -> fp32. transpose_store: C[n*ldc+m].
// act: 0 = none, 1 = exact GELU.
__global__ void __launch_bounds__(256)
gemm_bf16_wmma(const bf16_t* __restrict__ A, const bf16_t* __restrict__ W,
               const float* __restrict__ bias, void* __restrict__ Cout,
               int M, int N, int K, int lda, int ldw, int ldc,
               float alpha, int act, int store_bf16, int transpose_store)
{
    const int wave = threadIdx.x >> 5;
    const int lane = threadIdx.x & 31;
    const int l15  = lane & 15;
    const int hi   = lane >> 4;
    const int khi  = hi * 8;

    const int tilesN = N >> 5;
    const int tilesM = M >> 5;
    int tile = blockIdx.x * 8 + wave;
    if (tile >= tilesM * tilesN) return;
    int tm = tile / tilesN;
    int tn = tile % tilesN;

    const bf16_t* a0p = A + (size_t)(tm * 32 + l15) * lda;
    const bf16_t* a1p = a0p + (size_t)16 * lda;
    const bf16_t* b0p = W + (size_t)(tn * 32 + l15) * ldw;
    const bf16_t* b1p = b0p + (size_t)16 * ldw;

    v8f acc00 = {}, acc01 = {}, acc10 = {}, acc11 = {};

    for (int k0 = 0; k0 < K; k0 += 32) {
        v16bf a0 = load_frag(a0p + k0, khi);
        v16bf a1 = load_frag(a1p + k0, khi);
        v16bf b0 = load_frag(b0p + k0, khi);
        v16bf b1 = load_frag(b1p + k0, khi);
        acc00 = __builtin_amdgcn_wmma_f32_16x16x32_bf16(false, a0, false, b0,
                                                        (short)0, acc00, false, false);
        acc01 = __builtin_amdgcn_wmma_f32_16x16x32_bf16(false, a0, false, b1,
                                                        (short)0, acc01, false, false);
        acc10 = __builtin_amdgcn_wmma_f32_16x16x32_bf16(false, a1, false, b0,
                                                        (short)0, acc10, false, false);
        acc11 = __builtin_amdgcn_wmma_f32_16x16x32_bf16(false, a1, false, b1,
                                                        (short)0, acc11, false, false);
    }

#pragma unroll
    for (int mi = 0; mi < 2; ++mi) {
#pragma unroll
        for (int ni = 0; ni < 2; ++ni) {
            v8f acc = (mi == 0) ? ((ni == 0) ? acc00 : acc01)
                                : ((ni == 0) ? acc10 : acc11);
            int nn = tn * 32 + ni * 16 + l15;
            float bz = bias ? bias[nn] : 0.0f;
#pragma unroll
            for (int r = 0; r < 8; ++r) {
                int mm = tm * 32 + mi * 16 + hi * 8 + r;
                float v = acc[r] * alpha + bz;
                if (act == 1)
                    v = 0.5f * v * (1.0f + erff(v * 0.70710678118654752f));
                size_t idx = transpose_store ? ((size_t)nn * ldc + mm)
                                             : ((size_t)mm * ldc + nn);
                if (store_bf16) ((bf16_t*)Cout)[idx] = (bf16_t)v;
                else            ((float*)Cout)[idx]  = v;
            }
        }
    }
}

// ---------------- conversion / padding ----------------

__global__ void pad_convert_bf16(const float* __restrict__ src, bf16_t* __restrict__ dst,
                                 int srcR, int srcC, int dstC, int total)
{
    int i = blockIdx.x * 256 + threadIdx.x;
    if (i >= total) return;
    int r = i / dstC, c = i % dstC;
    float v = (r < srcR && c < srcC) ? src[(size_t)r * srcC + c] : 0.0f;
    dst[i] = (bf16_t)v;
}

__global__ void pad_copy_f32(const float* __restrict__ src, float* __restrict__ dst,
                             int n_src, int n_dst)
{
    int i = blockIdx.x * 256 + threadIdx.x;
    if (i < n_dst) dst[i] = (i < n_src) ? src[i] : 0.0f;
}

__global__ void zero_f32(float* p, int n)
{
    int i = blockIdx.x * 256 + threadIdx.x;
    if (i < n) p[i] = 0.0f;
}

// ---------------- residual + LayerNorm (row = 1024) ----------------

__global__ void __launch_bounds__(256)
resid_ln(const float* __restrict__ A, const float* __restrict__ B,
         const float* __restrict__ g, const float* __restrict__ b,
         float* __restrict__ outf, bf16_t* __restrict__ outb)
{
    __shared__ float sh[256];
    int row = blockIdx.x, tid = threadIdx.x;
    const float* ar = A + (size_t)row * 1024;
    const float* br = B + (size_t)row * 1024;
    float v[4];
    float s = 0.0f;
#pragma unroll
    for (int i = 0; i < 4; ++i) {
        int c = tid + i * 256;
        v[i] = ar[c] + br[c];
        s += v[i];
    }
    s = block_reduce_sum256(s, sh);
    float mean = s * (1.0f / 1024.0f);
    float ss = 0.0f;
#pragma unroll
    for (int i = 0; i < 4; ++i) { float d = v[i] - mean; ss += d * d; }
    ss = block_reduce_sum256(ss, sh);
    float rstd = rsqrtf(ss * (1.0f / 1024.0f) + 1e-5f);
#pragma unroll
    for (int i = 0; i < 4; ++i) {
        int c = tid + i * 256;
        float y = (v[i] - mean) * rstd * g[c] + b[c];
        outf[(size_t)row * 1024 + c] = y;
        if (outb) outb[(size_t)row * 1024 + c] = (bf16_t)y;
    }
}

// ---------------- softmax over a 2048-wide score row ----------------
// Writes bf16 probabilities, accumulates |sum(p)-1| for attn_unity.
__global__ void __launch_bounds__(256)
softmax_row(const float* __restrict__ S, bf16_t* __restrict__ P, float* dev_acc)
{
    __shared__ float sh[256];
    int row = blockIdx.x, tid = threadIdx.x;
    const float* sr = S + (size_t)row * 2048;
    float x[8];
    float mx = -3.4e38f;
#pragma unroll
    for (int i = 0; i < 8; ++i) { x[i] = sr[tid + i * 256]; mx = fmaxf(mx, x[i]); }
    mx = block_reduce_max256(mx, sh);
    float sum = 0.0f;
#pragma unroll
    for (int i = 0; i < 8; ++i) { x[i] = expf(x[i] - mx); sum += x[i]; }
    sum = block_reduce_sum256(sum, sh);
    float inv = 1.0f / sum;
    float chk = 0.0f;
#pragma unroll
    for (int i = 0; i < 8; ++i) {
        float p = x[i] * inv;
        chk += p;
        P[(size_t)row * 2048 + tid + i * 256] = (bf16_t)p;
    }
    chk = block_reduce_sum256(chk, sh);
    if (tid == 0) atomicAdd(dev_acc, fabsf(chk - 1.0f));
}

// ---------------- column means (plasticity pre/post) ----------------

__global__ void col_mean(const float* __restrict__ X, float* __restrict__ out,
                         int rows, int cols)
{
    int c = blockIdx.x * 256 + threadIdx.x;
    if (c >= cols) return;
    float s = 0.0f;
    for (int r = 0; r < rows; ++r) s += X[(size_t)r * cols + c];
    out[c] = s / (float)rows;
}

// ---------------- plasticity loss: ||tanh(outer(post,pre)/phi)-tgt||_F ----

__global__ void __launch_bounds__(256)
plast_loss_partial(const float* __restrict__ post, const float* __restrict__ pre,
                   const float* __restrict__ target_p, float* acc)
{
    __shared__ float sh[256];
    int idx = blockIdx.x * 256 + threadIdx.x;   // 1024*1024 total
    int i = idx >> 10, j = idx & 1023;
    float t = tanhf(post[i] * pre[j] * (1.0f / PHI_F));
    float d = t - *target_p;
    float v = block_reduce_sum256(d * d, sh);
    if (threadIdx.x == 0) atomicAdd(acc, v);
}

// ---------------- neuromorphic row statistics ----------------
// h = 0 => nh = total/tau_mem + homeo/tau_syn; spike reset; pop = row mean.
__global__ void __launch_bounds__(256)
neuro_row(const float* __restrict__ total,
          const float* tau_m, const float* tau_s, const float* gain,
          const float* attract, const float* vth, const float* vr,
          float* acc_sum, float* acc_sq)
{
    __shared__ float sh[256];
    int row = blockIdx.x, tid = threadIdx.x;
    float a_m = 1.0f / *tau_m;
    float a_s = 1.0f / *tau_s;
    float homeo = (*gain) * tanhf((*attract) * (1.0f / PHI_F));
    float th = *vth, rv = *vr;
    const float* tr = total + (size_t)row * 1024;
    float s = 0.0f;
#pragma unroll
    for (int i = 0; i < 4; ++i) {
        float nh = a_m * tr[tid + i * 256] + a_s * homeo;
        nh = (nh > th) ? rv : nh;
        s += nh;
    }
    s = block_reduce_sum256(s, sh);
    if (tid == 0) {
        float pop = s * (1.0f / 1024.0f);
        atomicAdd(acc_sum, pop);
        atomicAdd(acc_sq, pop * pop);
    }
}

// ---------------- finalize scalar outputs ----------------

__global__ void finalize_scalars(const float* scal, const float* ureg,
                                 const float* rate, const float* attract,
                                 float* out3)
{
    float dev = scal[0] * (1.0f / 65536.0f);            // mean over B*H*S rows
    out3[0] = expf(-(*ureg) * dev);                     // attn_unity
    out3[1] = sqrtf(scal[1]) * (*rate);                 // plast_loss
    float mean = scal[2] * (1.0f / 4096.0f);
    float var  = fmaxf(0.0f, scal[3] * (1.0f / 4096.0f) - mean * mean);
    out3[2] = expf(-var) * expf(-fabsf(mean - *attract));  // neuro_unity
}

// ---------------------------------------------------------------------------

extern "C" void kernel_launch(void* const* d_in, const int* in_sizes, int n_in,
                              void* d_out, int out_size, void* d_ws, size_t ws_size,
                              hipStream_t stream)
{
    (void)in_sizes; (void)n_in; (void)out_size; (void)ws_size;

    // -------- inputs (setup_inputs dict order) --------
    const float* x_in   = (const float*)d_in[0];
    const float* q_w    = (const float*)d_in[1];
    const float* k_w    = (const float*)d_in[2];
    const float* v_w    = (const float*)d_in[3];
    const float* o_w    = (const float*)d_in[4];
    const float* o_b    = (const float*)d_in[5];
    const float* ureg   = (const float*)d_in[6];
    const float* pl_w   = (const float*)d_in[7];
    const float* pl_b   = (const float*)d_in[8];
    const float* pl_rt  = (const float*)d_in[9];
    const float* u_tgt  = (const float*)d_in[10];
    const float* ci_w   = (const float*)d_in[11];
    const float* ci_b   = (const float*)d_in[12];
    /* cell_rec_w d_in[13] unused: h=0 -> tanh(h)=0 */
    const float* tau_m  = (const float*)d_in[14];
    const float* tau_s  = (const float*)d_in[15];
    const float* vth    = (const float*)d_in[16];
    const float* vrst   = (const float*)d_in[17];
    const float* hgain  = (const float*)d_in[18];
    const float* u_attr = (const float*)d_in[19];
    const float* w1     = (const float*)d_in[20];
    const float* b1     = (const float*)d_in[21];
    const float* w2     = (const float*)d_in[22];
    const float* b2     = (const float*)d_in[23];
    const float* ln1g   = (const float*)d_in[24];
    const float* ln1b   = (const float*)d_in[25];
    const float* ln2g   = (const float*)d_in[26];
    const float* ln2b   = (const float*)d_in[27];
    const float* ln3g   = (const float*)d_in[28];
    const float* ln3b   = (const float*)d_in[29];

    const int M4 = 4096, DD = 1024, SS = 2048, DFF = 1656, DFFP = 1664;

    // -------- workspace layout --------
    char*  base = (char*)d_ws;
    size_t off  = 0;
    auto take = [&](size_t bytes) -> void* {
        void* p = base + off;
        off = (off + bytes + 255) & ~(size_t)255;
        return p;
    };
    bf16_t* Xb   = (bf16_t*)take((size_t)M4 * DD * 2);       // current x (bf16)
    bf16_t* Qb   = (bf16_t*)take((size_t)M4 * DD * 2);
    bf16_t* Kb   = (bf16_t*)take((size_t)M4 * DD * 2);
    bf16_t* Vt   = (bf16_t*)take((size_t)DD * M4 * 2);       // V transposed [d][b*s]
    bf16_t* Pb   = (bf16_t*)take((size_t)SS * SS * 2);       // probs per (b,h)
    bf16_t* Cb   = (bf16_t*)take((size_t)M4 * DD * 2);       // ctx (bf16)
    bf16_t* Hb   = (bf16_t*)take((size_t)M4 * DFFP * 2);     // ffn hidden (bf16)
    bf16_t* qwb  = (bf16_t*)take((size_t)DD * DD * 2);
    bf16_t* kwb  = (bf16_t*)take((size_t)DD * DD * 2);
    bf16_t* vwb  = (bf16_t*)take((size_t)DD * DD * 2);
    bf16_t* owb  = (bf16_t*)take((size_t)DD * DD * 2);
    bf16_t* plwb = (bf16_t*)take((size_t)DD * DD * 2);
    bf16_t* ciwb = (bf16_t*)take((size_t)DD * DD * 2);
    bf16_t* w1p  = (bf16_t*)take((size_t)DFFP * DD * 2);     // zero-padded rows
    bf16_t* w2p  = (bf16_t*)take((size_t)DD * DFFP * 2);     // zero-padded cols
    float*  F0   = (float*)take((size_t)M4 * DD * 4);        // x2
    float*  F1   = (float*)take((size_t)M4 * DD * 4);        // x1
    float*  F2   = (float*)take((size_t)M4 * DD * 4);        // tmp out
    float*  SC   = (float*)take((size_t)SS * SS * 4);        // scores (reused)
    float*  b1p  = (float*)take((size_t)DFFP * 4);
    float*  scal = (float*)take((size_t)(16 + 2 * DD) * 4);  // accum + pre + post
    float*  pre  = scal + 16;
    float*  post = scal + 16 + DD;

    float* outx = (float*)d_out;
    float* out3 = outx + (size_t)M4 * DD;

    const float inv_scale = 1.0f / sqrtf(64.0f * PHI_F);

    // -------- stage 0: zero accumulators, convert activations + weights ----
    zero_f32<<<1, 256, 0, stream>>>(scal, 8);

    int nXD = M4 * DD;           // 4,194,304
    int nWW = DD * DD;           // 1,048,576
    pad_convert_bf16<<<nXD / 256, 256, 0, stream>>>(x_in, Xb, M4, DD, DD, nXD);
    pad_convert_bf16<<<nWW / 256, 256, 0, stream>>>(q_w,  qwb,  DD, DD, DD, nWW);
    pad_convert_bf16<<<nWW / 256, 256, 0, stream>>>(k_w,  kwb,  DD, DD, DD, nWW);
    pad_convert_bf16<<<nWW / 256, 256, 0, stream>>>(v_w,  vwb,  DD, DD, DD, nWW);
    pad_convert_bf16<<<nWW / 256, 256, 0, stream>>>(o_w,  owb,  DD, DD, DD, nWW);
    pad_convert_bf16<<<nWW / 256, 256, 0, stream>>>(pl_w, plwb, DD, DD, DD, nWW);
    pad_convert_bf16<<<nWW / 256, 256, 0, stream>>>(ci_w, ciwb, DD, DD, DD, nWW);
    int nW1 = DFFP * DD;         // 1,703,936
    pad_convert_bf16<<<nW1 / 256, 256, 0, stream>>>(w1, w1p, DFF, DD,  DD,   nW1);
    pad_convert_bf16<<<nW1 / 256, 256, 0, stream>>>(w2, w2p, DD,  DFF, DFFP, nW1);
    pad_copy_f32<<<(DFFP + 255) / 256, 256, 0, stream>>>(b1, b1p, DFF, DFFP);

    // -------- stage 1: Q, K, V projections (WMMA) --------
    // tiles = (4096/32)*(1024/32) = 4096 -> 512 blocks of 8 waves
    gemm_bf16_wmma<<<512, 256, 0, stream>>>(Xb, qwb, nullptr, Qb,
        M4, DD, DD, DD, DD, DD, 1.0f, 0, /*bf16*/1, /*T*/0);
    gemm_bf16_wmma<<<512, 256, 0, stream>>>(Xb, kwb, nullptr, Kb,
        M4, DD, DD, DD, DD, DD, 1.0f, 0, 1, 0);
    gemm_bf16_wmma<<<512, 256, 0, stream>>>(Xb, vwb, nullptr, Vt,
        M4, DD, DD, DD, DD, /*ldc=*/M4, 1.0f, 0, 1, /*T*/1);   // Vt[d][b*s]

    // -------- stage 2: attention per (b,h), score buffer reused --------
    for (int b = 0; b < 2; ++b) {
        for (int h = 0; h < 16; ++h) {
            const bf16_t* Qh = Qb + (size_t)b * SS * DD + h * 64;
            const bf16_t* Kh = Kb + (size_t)b * SS * DD + h * 64;
            // scores = (Q_bh @ K_bh^T) / sqrt(HD*phi) : 2048x2048, K=64
            gemm_bf16_wmma<<<512, 256, 0, stream>>>(Qh, Kh, nullptr, SC,
                SS, SS, 64, DD, DD, SS, inv_scale, 0, /*f32*/0, 0);
            softmax_row<<<SS, 256, 0, stream>>>(SC, Pb, &scal[0]);
            // ctx = P @ V_bh : 2048x64, K=2048 (V supplied transposed)
            const bf16_t* Vh = Vt + (size_t)h * 64 * M4 + (size_t)b * SS;
            bf16_t* Ch = Cb + (size_t)b * SS * DD + h * 64;
            gemm_bf16_wmma<<<16, 256, 0, stream>>>(Pb, Vh, nullptr, Ch,
                SS, 64, SS, SS, M4, DD, 1.0f, 0, /*bf16*/1, 0);
        }
    }

    // -------- stage 3: output projection + LN1 --------
    gemm_bf16_wmma<<<512, 256, 0, stream>>>(Cb, owb, o_b, F2,
        M4, DD, DD, DD, DD, DD, 1.0f, 0, /*f32*/0, 0);
    resid_ln<<<M4, 256, 0, stream>>>(x_in, F2, ln1g, ln1b, F1, Xb); // x1

    // -------- stage 4: plasticity --------
    gemm_bf16_wmma<<<512, 256, 0, stream>>>(Xb, plwb, pl_b, F2,
        M4, DD, DD, DD, DD, DD, 1.0f, 0, 0, 0);                    // plast_out
    col_mean<<<DD / 256, 256, 0, stream>>>(F1, pre,  M4, DD);
    col_mean<<<DD / 256, 256, 0, stream>>>(F2, post, M4, DD);
    plast_loss_partial<<<(DD * DD) / 256, 256, 0, stream>>>(post, pre, u_tgt, &scal[1]);
    resid_ln<<<M4, 256, 0, stream>>>(F1, F2, ln2g, ln2b, F0, Xb);  // x2

    // -------- stage 5: neuromorphic unity (cell input GEMM + row stats) ----
    gemm_bf16_wmma<<<512, 256, 0, stream>>>(Xb, ciwb, ci_b, F2,
        M4, DD, DD, DD, DD, DD, 1.0f, 0, 0, 0);
    neuro_row<<<M4, 256, 0, stream>>>(F2, tau_m, tau_s, hgain, u_attr,
                                      vth, vrst, &scal[2], &scal[3]);

    // -------- stage 6: FFN (padded to 1664) + LN3 --------
    // tiles = 128 * (1664/32=52) = 6656 -> 832 blocks
    gemm_bf16_wmma<<<832, 256, 0, stream>>>(Xb, w1p, b1p, Hb,
        M4, DFFP, DD, DD, DD, DFFP, 1.0f, /*gelu*/1, /*bf16*/1, 0);
    gemm_bf16_wmma<<<512, 256, 0, stream>>>(Hb, w2p, b2, F2,
        M4, DD, DFFP, DFFP, DFFP, DD, 1.0f, 0, 0, 0);
    resid_ln<<<M4, 256, 0, stream>>>(F0, F2, ln3g, ln3b, outx, nullptr);

    // -------- stage 7: scalar outputs --------
    finalize_scalars<<<1, 1, 0, stream>>>(scal, ureg, pl_rt, u_attr, out3);
}
